// LocallyGroupedAttn_66279935311967
// MI455X (gfx1250) — matrix operand order repaired and checked by reading, not compiled, not run on hardware
//
#include <hip/hip_runtime.h>

typedef __attribute__((ext_vector_type(16))) __bf16 v16bf;
typedef __attribute__((ext_vector_type(8)))  __bf16 v8bf;
typedef __attribute__((ext_vector_type(8)))  float  v8f;

static __device__ __forceinline__ unsigned short bfbits(float f) {
    unsigned u = __builtin_bit_cast(unsigned, f);
    u += 0x7fffu + ((u >> 16) & 1u);          // round-to-nearest-even
    return (unsigned short)(u >> 16);
}
static __device__ __forceinline__ __bf16 f2bf(float f) {
    unsigned short h = bfbits(f);
    return __builtin_bit_cast(__bf16, h);
}

union Frag { v16bf v; v8bf h[2]; };
// Build a 16x(bf16) fragment from two 16-byte chunks (ds_load_b128 / global_load_b128)
static __device__ __forceinline__ v16bf ld_frag(const __bf16* p0, const __bf16* p1) {
    Frag u;
    u.h[0] = *(const v8bf*)p0;
    u.h[1] = *(const v8bf*)p1;
    return u.v;
}

static __device__ __forceinline__ v8f wmma_bf16(v16bf a, v16bf b, v8f c) {
    return __builtin_amdgcn_wmma_f32_16x16x32_bf16(false, a, false, b, (short)0, c, false, false);
}

// ---- problem constants ----
#define CDIM   256
#define HWIMG  112
#define NPOS   12544            // 112*112
#define NH     8
#define HD     32
#define WS7    7

// ---- LDS layout (units: __bf16 halves). Row pads of 8 halves stagger banks. ----
#define QK_STRIDE 520           // 512 + 8   : [64 tokens][Q(256) K(256)] row-major
#define VT_STRIDE 72            // 64 + 8    : [256 chans][64 tokens]  (V transposed)
#define XO_STRIDE 264           // 256 + 8   : [64 tokens][256 chans]  (x, then O)
#define P_STRIDE  72            // 64 + 8    : per head [64 rows][64 keys]
#define OFF_QK 0
#define OFF_VT (OFF_QK + 64 * QK_STRIDE)        // 33280
#define OFF_XO (OFF_VT + 256 * VT_STRIDE)       // 51712
#define OFF_P  (OFF_XO + 64 * XO_STRIDE)        // 68608
#define SMEM_HALVES (OFF_P + NH * 64 * P_STRIDE) // 105472 halves = 210944 B
#define SMEM_BYTES  (SMEM_HALVES * 2)

// ------------------------------------------------------------------
// Kernel 0: one-time fp32 -> bf16 weight conversion into workspace
// ------------------------------------------------------------------
__global__ void lga_wconv(const float* __restrict__ qw, const float* __restrict__ pw,
                          __bf16* __restrict__ oq, __bf16* __restrict__ op) {
    int i = blockIdx.x * 256 + threadIdx.x;
    if (i < 768 * 256)  oq[i] = f2bf(qw[i]);
    if (i < 256 * 256)  op[i] = f2bf(pw[i]);
}

// ------------------------------------------------------------------
// Kernel 1: fully fused window attention. 1 block = 1 window (49 tokens).
// ------------------------------------------------------------------
__global__ __launch_bounds__(256, 1)
void lga_fused(const float* __restrict__ x,
               const float* __restrict__ qkv_b,
               const float* __restrict__ proj_b,
               const __bf16* __restrict__ wq,   // [768][256] bf16
               const __bf16* __restrict__ wp,   // [256][256] bf16
               float* __restrict__ out) {
    extern __shared__ __bf16 smem[];
    __bf16* sQK = smem + OFF_QK;
    __bf16* sVT = smem + OFF_VT;
    __bf16* sXO = smem + OFF_XO;   // x (stage1) then O (stage2/3)
    __bf16* sP  = smem + OFF_P;

    const int tid  = threadIdx.x;
    const int wv   = tid >> 5;     // wave id 0..7
    const int lane = tid & 31;
    const int hl   = lane >> 4;    // half-wave select
    const int nloc = lane & 15;

    const int blk = blockIdx.x;
    const int b   = blk >> 8;      // batch
    const int win = blk & 255;
    const int bh  = win >> 4;
    const int bw  = win & 15;

    // ---------------- Stage 1a: load x window -> LDS bf16 (pad rows zeroed) ----
    {
        const int row = tid >> 2;
        const int ch0 = (tid & 3) * 64;
        __bf16* xr = sXO + row * XO_STRIDE + ch0;
        if (row < 49) {
            const int i = row / WS7, j = row % WS7;
            const float* src = x + (size_t)(b * NPOS + (bh * WS7 + i) * HWIMG + (bw * WS7 + j)) * CDIM + ch0;
            #pragma unroll
            for (int c = 0; c < 16; ++c) {
                float4 v = *(const float4*)(src + c * 4);
                unsigned long long pk = (unsigned long long)bfbits(v.x)
                                      | ((unsigned long long)bfbits(v.y) << 16)
                                      | ((unsigned long long)bfbits(v.z) << 32)
                                      | ((unsigned long long)bfbits(v.w) << 48);
                *(unsigned long long*)(xr + c * 4) = pk;
            }
        } else {
            #pragma unroll
            for (int c = 0; c < 16; ++c)
                *(unsigned long long*)(xr + c * 4) = 0ull;
        }
    }
    __syncthreads();

    // ---------------- Stage 1b: QKV GEMM  [64 x 256] @ [256 x 768] ------------
    // wave wv owns N-tile columns {wv, wv+8, ..., wv+40}
    for (int nc = 0; nc < 6; ++nc) {
        const int nbase = (wv + nc * 8) * 16;
        v8f acc[4] = {};
        const __bf16* wrow = wq + (size_t)(nbase + nloc) * CDIM;
        #pragma unroll
        for (int ks = 0; ks < 8; ++ks) {
            const int kb = ks * 32;
            const __bf16* bp = wrow + kb + 8 * hl;
            v16bf bfr = ld_frag(bp, bp + 16);
            #pragma unroll
            for (int mt = 0; mt < 4; ++mt) {
                const __bf16* ap = sXO + (mt * 16 + nloc) * XO_STRIDE + kb + 8 * hl;
                v16bf afr = ld_frag(ap, ap + 16);
                acc[mt] = wmma_bf16(afr, bfr, acc[mt]);
            }
        }
        const int   n    = nbase + nloc;
        const float bias = qkv_b[n];
        #pragma unroll
        for (int mt = 0; mt < 4; ++mt) {
            #pragma unroll
            for (int r = 0; r < 8; ++r) {
                const int m = mt * 16 + r + 8 * hl;
                const float val = acc[mt][r] + bias;
                if (n < 512) sQK[m * QK_STRIDE + n] = f2bf(val);          // Q,K row-major
                else         sVT[(n - 512) * VT_STRIDE + m] = f2bf(val);  // V transposed
            }
        }
    }
    __syncthreads();

    // ---------------- Stage 2: attention, one wave per head -------------------
    {
        const int h = wv;
        // cache the 4 Q A-fragments (K-dim = hd = 32 -> single k-step)
        v16bf qf[4];
        #pragma unroll
        for (int mt = 0; mt < 4; ++mt) {
            const __bf16* p = sQK + (mt * 16 + nloc) * QK_STRIDE + h * HD + 8 * hl;
            qf[mt] = ld_frag(p, p + 16);
        }
        v8f sacc[4][4] = {};
        #pragma unroll
        for (int nt = 0; nt < 4; ++nt) {
            const __bf16* p = sQK + (nt * 16 + nloc) * QK_STRIDE + 256 + h * HD + 8 * hl;
            v16bf kf = ld_frag(p, p + 16);
            #pragma unroll
            for (int mt = 0; mt < 4; ++mt)
                sacc[mt][nt] = wmma_bf16(qf[mt], kf, sacc[mt][nt]);
        }

        // in-register softmax over keys (cols), masking keys >= 49
        __bf16* Ph = sP + h * (64 * P_STRIDE);
        const float scale = 0.17677669529663687f;    // 32^-0.5
        #pragma unroll
        for (int mt = 0; mt < 4; ++mt) {
            #pragma unroll
            for (int r = 0; r < 8; ++r) {
                float s0 = sacc[mt][0][r] * scale;
                float s1 = sacc[mt][1][r] * scale;
                float s2 = sacc[mt][2][r] * scale;
                float s3 = sacc[mt][3][r] * scale;
                const bool m3 = (nloc >= 1);         // key index 48+nloc >= 49
                float mx = fmaxf(fmaxf(s0, s1), fmaxf(s2, m3 ? -3.0e38f : s3));
                mx = fmaxf(mx, __shfl_xor(mx, 1, 32));
                mx = fmaxf(mx, __shfl_xor(mx, 2, 32));
                mx = fmaxf(mx, __shfl_xor(mx, 4, 32));
                mx = fmaxf(mx, __shfl_xor(mx, 8, 32));   // reduced over 16-lane half
                float e0 = __expf(s0 - mx);
                float e1 = __expf(s1 - mx);
                float e2 = __expf(s2 - mx);
                float e3 = m3 ? 0.0f : __expf(s3 - mx);
                float sm = e0 + e1 + e2 + e3;
                sm += __shfl_xor(sm, 1, 32);
                sm += __shfl_xor(sm, 2, 32);
                sm += __shfl_xor(sm, 4, 32);
                sm += __shfl_xor(sm, 8, 32);
                const float inv = 1.0f / sm;
                const int m = mt * 16 + r + 8 * hl;
                __bf16* pr = Ph + m * P_STRIDE + nloc;
                pr[0]  = f2bf(e0 * inv);
                pr[16] = f2bf(e1 * inv);
                pr[32] = f2bf(e2 * inv);
                pr[48] = f2bf(e3 * inv);
            }
        }

        // O = P @ V   (K-dim 64 -> 2 k-steps; N = hd = 32 -> 2 N-tiles)
        v8f oacc[4][2] = {};
        #pragma unroll
        for (int ks = 0; ks < 2; ++ks) {
            const int kb = ks * 32;
            #pragma unroll
            for (int nt = 0; nt < 2; ++nt) {
                const __bf16* bp = sVT + (h * HD + nt * 16 + nloc) * VT_STRIDE + kb + 8 * hl;
                v16bf vf = ld_frag(bp, bp + 16);
                #pragma unroll
                for (int mt = 0; mt < 4; ++mt) {
                    const __bf16* ap = Ph + (mt * 16 + nloc) * P_STRIDE + kb + 8 * hl;
                    v16bf pf = ld_frag(ap, ap + 16);
                    oacc[mt][nt] = wmma_bf16(pf, vf, oacc[mt][nt]);
                }
            }
        }
        #pragma unroll
        for (int mt = 0; mt < 4; ++mt)
            #pragma unroll
            for (int nt = 0; nt < 2; ++nt)
                #pragma unroll
                for (int r = 0; r < 8; ++r) {
                    const int m = mt * 16 + r + 8 * hl;
                    sXO[m * XO_STRIDE + h * HD + nt * 16 + nloc] = f2bf(oacc[mt][nt][r]);
                }
    }
    __syncthreads();

    // ---------------- Stage 3: proj GEMM + bias, scatter to global -------------
    #pragma unroll
    for (int nc = 0; nc < 2; ++nc) {
        const int nbase = (wv + nc * 8) * 16;
        v8f acc[4] = {};
        const __bf16* wrow = wp + (size_t)(nbase + nloc) * CDIM;
        #pragma unroll
        for (int ks = 0; ks < 8; ++ks) {
            const int kb = ks * 32;
            const __bf16* bp = wrow + kb + 8 * hl;
            v16bf bfr = ld_frag(bp, bp + 16);
            #pragma unroll
            for (int mt = 0; mt < 4; ++mt) {
                const __bf16* ap = sXO + (mt * 16 + nloc) * XO_STRIDE + kb + 8 * hl;
                v16bf afr = ld_frag(ap, ap + 16);
                acc[mt] = wmma_bf16(afr, bfr, acc[mt]);
            }
        }
        const float bias = proj_b[nbase + nloc];
        #pragma unroll
        for (int mt = 0; mt < 4; ++mt) {
            #pragma unroll
            for (int r = 0; r < 8; ++r) {
                const int m = mt * 16 + r + 8 * hl;
                if (m < 49) {
                    const int i = m / WS7, j = m % WS7;
                    const size_t ridx = (size_t)(b * NPOS + (bh * WS7 + i) * HWIMG + (bw * WS7 + j));
                    out[ridx * CDIM + nbase + nloc] = acc[mt][r] + bias;
                }
            }
        }
    }
}

extern "C" void kernel_launch(void* const* d_in, const int* in_sizes, int n_in,
                              void* d_out, int out_size, void* d_ws, size_t ws_size,
                              hipStream_t stream) {
    (void)in_sizes; (void)n_in; (void)out_size; (void)ws_size;
    const float* x      = (const float*)d_in[0];
    const float* qkv_w  = (const float*)d_in[1];
    const float* qkv_b  = (const float*)d_in[2];
    const float* proj_w = (const float*)d_in[3];
    const float* proj_b = (const float*)d_in[4];
    float* out = (float*)d_out;

    __bf16* wq = (__bf16*)d_ws;            // 768*256 bf16
    __bf16* wp = wq + 768 * 256;           // 256*256 bf16

    lga_wconv<<<768, 256, 0, stream>>>(qkv_w, proj_w, wq, wp);
    lga_fused<<<2048, 256, SMEM_BYTES, stream>>>(x, qkv_b, proj_b, wq, wp, out);
}
